// LMKT_23115513987496
// MI455X (gfx1250) — compile-verified, matrix operand order checked by reading, not run on hardware
//
#include <hip/hip_runtime.h>
#include <hip/hip_bf16.h>

// ---------------------------------------------------------------------------
// Types for CDNA5 WMMA (wave32): D(f32 16x16) = A(f16 16x32) x B(f16 32x16) + C
// ---------------------------------------------------------------------------
typedef _Float16 v8h  __attribute__((ext_vector_type(8)));
typedef _Float16 v16h __attribute__((ext_vector_type(16)));
typedef float    v8f  __attribute__((ext_vector_type(8)));

// Per ISA 16-bit A/B layout: for lane l (m = l&15, half = l>>4) the 16 halves
// of the fragment are K = {k0 + half*8 + [0..7], k0 + half*8 + 16 + [0..7]}.
// Caller passes base = row_ptr + k0 + half*8; we do two 16-byte loads.
__device__ inline v16h load_frag(const _Float16* __restrict__ p) {
    v8h lo = *(const v8h*)(p);
    v8h hi = *(const v8h*)(p + 16);
    v16h r;
#pragma unroll
    for (int i = 0; i < 8; ++i) { r[i] = lo[i]; r[i + 8] = hi[i]; }
    return r;
}

__device__ inline v8f wmma_f16(v16h a, v16h b, v8f c) {
    return __builtin_amdgcn_wmma_f32_16x16x32_f16(
        /*neg_a=*/false, a, /*neg_b=*/false, b,
        /*c_mod=*/(short)0, c, /*reuse_a=*/false, /*reuse_b=*/false);
}

// ---------------------------------------------------------------------------
// Elementwise converts / transposes
// ---------------------------------------------------------------------------
__global__ void cvt_f32_to_f16(const float* __restrict__ in,
                               _Float16* __restrict__ out, int n) {
    int i = blockIdx.x * 256 + threadIdx.x;
    if (i < n) out[i] = (_Float16)in[i];
}

// W [K,N] f32 -> Wt [N,K] f16   (weight transpose+downconvert)
__global__ void transpose_w(const float* __restrict__ W,
                            _Float16* __restrict__ Wt, int K, int N) {
    int i = blockIdx.x * 256 + threadIdx.x;
    if (i < K * N) {
        int k = i / N, n = i % N;
        Wt[(size_t)n * K + k] = (_Float16)W[i];
    }
}

// Vrow [B*S, D] f16 (heads packed in D) -> Vt [B,H,DK,S] f16
__global__ void transpose_v(const _Float16* __restrict__ Vr,
                            _Float16* __restrict__ Vt,
                            int Bn, int S, int H, int DK) {
    int D = H * DK;
    int total = Bn * S * D;
    int i = blockIdx.x * 256 + threadIdx.x;
    if (i < total) {
        int d  = i % D;
        int bs = i / D;
        int s  = bs % S;
        int b  = bs / S;
        int h  = d / DK, dk = d % DK;
        Vt[(((size_t)b * H + h) * DK + dk) * S + s] = Vr[i];
    }
}

// ---------------------------------------------------------------------------
// Register-blocked WMMA GEMM: one wave computes a 16x64 output tile
// (4 N-subtiles sharing one A fragment per K-step -> 4 independent WMMA
// chains, 4x less A traffic). out = A[M,K]*Bt[N,K]^T + bias (+resid)(+relu).
// Requires N % 64 == 0.
// ---------------------------------------------------------------------------
__global__ __launch_bounds__(32)
void gemm16x64(const _Float16* __restrict__ A, const _Float16* __restrict__ Bt,
               const float* __restrict__ bias, const float* __restrict__ resid,
               float* __restrict__ outF, _Float16* __restrict__ outH,
               int M, int N, int K, int relu) {
    int nblk = N >> 6;                       // 64-wide column blocks
    int tm = blockIdx.x / nblk;
    int tb = blockIdx.x % nblk;
    int lane = threadIdx.x;
    int m = lane & 15, half = lane >> 4;

    const _Float16* arow = A + (size_t)(tm * 16 + m) * K + half * 8;
    const _Float16* brow[4];
#pragma unroll
    for (int t = 0; t < 4; ++t)
        brow[t] = Bt + (size_t)(tb * 64 + t * 16 + m) * K + half * 8;

    v8f acc[4] = {};
    for (int k0 = 0; k0 < K; k0 += 32) {
        if (k0 + 64 <= K) {                  // keep next K-slab in near caches
            __builtin_prefetch(arow + k0 + 32, 0, 3);
#pragma unroll
            for (int t = 0; t < 4; ++t)
                __builtin_prefetch(brow[t] + k0 + 32, 0, 3);
        }
        v16h a = load_frag(arow + k0);
#pragma unroll
        for (int t = 0; t < 4; ++t) {
            v16h b = load_frag(brow[t] + k0);
            acc[t] = wmma_f16(a, b, acc[t]);
        }
    }

    int row0 = tm * 16 + half * 8;
#pragma unroll
    for (int t = 0; t < 4; ++t) {
        int col = tb * 64 + t * 16 + m;
        float bv = bias ? bias[col] : 0.0f;
#pragma unroll
        for (int r = 0; r < 8; ++r) {
            int row = row0 + r;
            float v = acc[t][r] + bv;
            if (relu) v = fmaxf(v, 0.0f);
            size_t idx = (size_t)row * N + col;
            if (resid) v += resid[idx];
            if (outF) outF[idx] = v;
            if (outH) outH[idx] = (_Float16)v;
        }
    }
}

// ---------------------------------------------------------------------------
// Attention: one wave per (b, h, 16-row query tile).
// Phase 1: scores = Q Kt / 8 via WMMA -> LDS (16 x 512 f32)
// Phase 2: masked softmax -> cumsum -> distance decay -> softmax (1 lane/row)
// Phase 3: ctx = attn @ V via WMMA, A-frag from LDS shared by 4 accumulators
// ---------------------------------------------------------------------------
__global__ __launch_bounds__(32)
void attn_kernel(const _Float16* __restrict__ Q, const _Float16* __restrict__ Kh,
                 const _Float16* __restrict__ Vt,
                 const float* __restrict__ gammas, const int* __restrict__ maskp,
                 _Float16* __restrict__ ctx,
                 int Bn, int S, int H, int DK) {
    __shared__ float sc[16][512];
    __shared__ __align__(16) _Float16 at[16][512];

    const int D = H * DK;
    int t  = blockIdx.x;
    int nqt = S >> 4;                  // 32
    int qt = t % nqt;
    int h  = (t / nqt) % H;
    int b  = t / (nqt * H);

    int lane = threadIdx.x;
    int m = lane & 15, half = lane >> 4;
    int mask_val = maskp[0];

    // --- Phase 1: scores tile (16 x S) ---
    const _Float16* qrow = Q + ((size_t)(b * S + qt * 16 + m) * D) + h * DK + half * 8;
    v16h qa0 = load_frag(qrow);
    v16h qa1 = load_frag(qrow + 32);

    for (int nt = 0; nt < (S >> 4); ++nt) {
        const _Float16* krow =
            Kh + ((size_t)(b * S + nt * 16 + m) * D) + h * DK + half * 8;
        v16h kb0 = load_frag(krow);
        v16h kb1 = load_frag(krow + 32);
        v8f acc = {};
        acc = wmma_f16(qa0, kb0, acc);
        acc = wmma_f16(qa1, kb1, acc);
#pragma unroll
        for (int r = 0; r < 8; ++r)
            sc[half * 8 + r][nt * 16 + m] = acc[r] * 0.125f;  // 1/sqrt(64)
    }
    __syncthreads();

    // --- Phase 2: per-row softmax + distance-decay + softmax ---
    if (lane < 16) {
        int i = qt * 16 + lane;                       // global query index
        int len = i + mask_val;                       // allowed j: j - i < mask
        if (len > S) len = S;
        if (len < 0) len = 0;
        float*     row = sc[lane];
        _Float16*  ar  = at[lane];

        float mx = -3.0e38f;
        for (int j = 0; j < len; ++j) mx = fmaxf(mx, row[j]);
        float sum = 0.0f;
        for (int j = 0; j < len; ++j) sum += expf(row[j] - mx);
        float inv = (sum > 0.0f) ? 1.0f / sum : 0.0f;

        float g  = gammas[h];
        float sp = (g > 20.0f) ? g : log1pf(expf(g));  // softplus
        float gamma = -sp;

        float cum = 0.0f, mx2 = -3.0e38f;
        for (int j = 0; j < len; ++j) {
            float p = expf(row[j] - mx) * inv;        // scores_ (sums to 1)
            cum += p;                                  // distcum (inclusive)
            float pe = fabsf((float)(i - j));          // position_effect
            float ds = sqrtf(fmaxf((1.0f - cum) * pe, 0.0f));
            float te = expf(ds * gamma);
            te = fminf(fmaxf(te, 1e-5f), 1e5f);        // total_effect clamp
            float s2 = row[j] * te;
            row[j] = s2;
            mx2 = fmaxf(mx2, s2);
        }
        float sum2 = 0.0f;
        for (int j = 0; j < len; ++j) sum2 += expf(row[j] - mx2);
        float inv2 = (sum2 > 0.0f) ? 1.0f / sum2 : 0.0f;
        for (int j = 0; j < len; ++j)
            ar[j] = (_Float16)(expf(row[j] - mx2) * inv2);
        for (int j = len; j < S; ++j) ar[j] = (_Float16)0.0f;  // zero-pad for WMMA K
    }
    __syncthreads();

    // --- Phase 3: ctx(16 x 64) = attn(16 x S) @ V(S x 64), 4 acc chains ---
    const _Float16* vbase =
        Vt + ((size_t)((b * H + h) * DK) + m) * S + half * 8;
    v8f acc[4] = {};
    for (int kt = 0; kt < (S >> 5); ++kt) {
        v16h a = load_frag(&at[m][kt * 32 + half * 8]);   // ds_load_b128
#pragma unroll
        for (int nt = 0; nt < 4; ++nt) {
            v16h v = load_frag(vbase + (size_t)(nt * 16) * S + kt * 32);
            acc[nt] = wmma_f16(a, v, acc[nt]);
        }
    }
#pragma unroll
    for (int nt = 0; nt < 4; ++nt) {
#pragma unroll
        for (int r = 0; r < 8; ++r) {
            int srow = qt * 16 + half * 8 + r;
            ctx[((size_t)(b * S + srow)) * D + h * DK + nt * 16 + m] =
                (_Float16)acc[nt][r];
        }
    }
}

// ---------------------------------------------------------------------------
// Row LayerNorm: one block (256 threads) per row of D elements.
// ---------------------------------------------------------------------------
__global__ __launch_bounds__(256)
void ln_kernel(const float* __restrict__ X, const float* __restrict__ g,
               const float* __restrict__ bta, float* __restrict__ outF,
               _Float16* __restrict__ outH, int D) {
    __shared__ float red[256];
    int row = blockIdx.x, tid = threadIdx.x;
    const float* x = X + (size_t)row * D;

    float s = 0.0f;
    for (int c = tid; c < D; c += 256) s += x[c];
    red[tid] = s; __syncthreads();
    for (int w = 128; w > 0; w >>= 1) {
        if (tid < w) red[tid] += red[tid + w];
        __syncthreads();
    }
    float mu = red[0] / (float)D;
    __syncthreads();

    float s2 = 0.0f;
    for (int c = tid; c < D; c += 256) { float d = x[c] - mu; s2 += d * d; }
    red[tid] = s2; __syncthreads();
    for (int w = 128; w > 0; w >>= 1) {
        if (tid < w) red[tid] += red[tid + w];
        __syncthreads();
    }
    float var = red[0] / (float)D;
    float rs = rsqrtf(var + 1e-5f);

    for (int c = tid; c < D; c += 256) {
        float v = (x[c] - mu) * rs * g[c] + bta[c];
        size_t idx = (size_t)row * D + c;
        if (outF) outF[idx] = v;
        if (outH) outH[idx] = (_Float16)v;
    }
}

// ---------------------------------------------------------------------------
// Orchestration
// ---------------------------------------------------------------------------
extern "C" void kernel_launch(void* const* d_in, const int* in_sizes, int n_in,
                              void* d_out, int out_size, void* d_ws, size_t ws_size,
                              hipStream_t stream) {
    (void)in_sizes; (void)n_in; (void)out_size; (void)ws_size;
    const int B = 16, S = 512, D = 512, H = 8, DFF = 2048, DK = D / H;
    const int M = B * S;  // 8192

    const float* query  = (const float*)d_in[0];
    const float* key    = (const float*)d_in[1];
    const float* values = (const float*)d_in[2];
    const float* Wk     = (const float*)d_in[3];
    const float* bk     = (const float*)d_in[4];
    const float* Wv     = (const float*)d_in[5];
    const float* bv     = (const float*)d_in[6];
    const float* Wo     = (const float*)d_in[7];
    const float* bo     = (const float*)d_in[8];
    const float* gammas = (const float*)d_in[9];
    const float* ln1_g  = (const float*)d_in[10];
    const float* ln1_b  = (const float*)d_in[11];
    const float* W1     = (const float*)d_in[12];
    const float* b1     = (const float*)d_in[13];
    const float* W2     = (const float*)d_in[14];
    const float* b2     = (const float*)d_in[15];
    const float* ln2_g  = (const float*)d_in[16];
    const float* ln2_b  = (const float*)d_in[17];
    const int*   maskp  = (const int*)d_in[18];

    char* ws = (char*)d_ws;
    size_t off = 0;
    auto take = [&](size_t bytes) -> char* {
        char* p = ws + off;
        off += (bytes + 255) & ~(size_t)255;
        return p;
    };

    _Float16* qh   = (_Float16*)take((size_t)M * D * 2);
    _Float16* kh   = (_Float16*)take((size_t)M * D * 2);
    _Float16* vh   = (_Float16*)take((size_t)M * D * 2);
    _Float16* wkt  = (_Float16*)take((size_t)D * D * 2);
    _Float16* wvt  = (_Float16*)take((size_t)D * D * 2);
    _Float16* wot  = (_Float16*)take((size_t)D * D * 2);
    _Float16* w1t  = (_Float16*)take((size_t)D * DFF * 2);
    _Float16* w2t  = (_Float16*)take((size_t)DFF * D * 2);
    _Float16* Qh   = (_Float16*)take((size_t)M * D * 2);   // 4x8MB contiguous
    _Float16* Kh   = (_Float16*)take((size_t)M * D * 2);
    _Float16* Vrow = (_Float16*)take((size_t)M * D * 2);
    _Float16* Vt   = (_Float16*)take((size_t)M * D * 2);
    _Float16* ctx  = (_Float16*)take((size_t)M * D * 2);
    float*    y1   = (float*)take((size_t)M * D * 4);
    float*    x1   = (float*)take((size_t)M * D * 4);
    _Float16* x1h  = (_Float16*)take((size_t)M * D * 2);
    _Float16* hh   = (_Float16*)Qh;   // reuse Qh..Vt (32 MB), dead after attn
    float*    y2   = y1;              // reuse y1, dead after LN1

    const int nMD  = M * D;
    dim3 blk(256);

    // f32 -> f16 activations
    cvt_f32_to_f16<<<(nMD + 255) / 256, blk, 0, stream>>>(query,  qh, nMD);
    cvt_f32_to_f16<<<(nMD + 255) / 256, blk, 0, stream>>>(key,    kh, nMD);
    cvt_f32_to_f16<<<(nMD + 255) / 256, blk, 0, stream>>>(values, vh, nMD);

    // weight transposes to [N,K] f16
    transpose_w<<<(D * D + 255) / 256, blk, 0, stream>>>(Wk, wkt, D, D);
    transpose_w<<<(D * D + 255) / 256, blk, 0, stream>>>(Wv, wvt, D, D);
    transpose_w<<<(D * D + 255) / 256, blk, 0, stream>>>(Wo, wot, D, D);
    transpose_w<<<(D * DFF + 255) / 256, blk, 0, stream>>>(W1, w1t, D, DFF);
    transpose_w<<<(DFF * D + 255) / 256, blk, 0, stream>>>(W2, w2t, DFF, D);

    // QKV projections (kq_same: Q and K both use Wk/bk)
    int gDD = (M / 16) * (D / 64);
    gemm16x64<<<gDD, 32, 0, stream>>>(qh, wkt, bk, nullptr, nullptr, Qh,   M, D, D, 0);
    gemm16x64<<<gDD, 32, 0, stream>>>(kh, wkt, bk, nullptr, nullptr, Kh,   M, D, D, 0);
    gemm16x64<<<gDD, 32, 0, stream>>>(vh, wvt, bv, nullptr, nullptr, Vrow, M, D, D, 0);

    transpose_v<<<(nMD + 255) / 256, blk, 0, stream>>>(Vrow, Vt, B, S, H, DK);

    // attention with distance decay
    attn_kernel<<<B * H * (S / 16), 32, 0, stream>>>(Qh, Kh, Vt, gammas, maskp,
                                                     ctx, B, S, H, DK);

    // out-proj + residual(query) -> y1, then LN1 -> x1 (f32) + x1h (f16)
    gemm16x64<<<gDD, 32, 0, stream>>>(ctx, wot, bo, query, y1, nullptr, M, D, D, 0);
    ln_kernel<<<M, 256, 0, stream>>>(y1, ln1_g, ln1_b, x1, x1h, D);

    // FFN
    int gDF = (M / 16) * (DFF / 64);
    gemm16x64<<<gDF, 32, 0, stream>>>(x1h, w1t, b1, nullptr, nullptr, hh, M, DFF, D, 1);
    gemm16x64<<<gDD, 32, 0, stream>>>(hh, w2t, b2, x1, y2, nullptr, M, D, DFF, 0);

    // LN2 -> final output (f32)
    ln_kernel<<<M, 256, 0, stream>>>(y2, ln2_g, ln2_b, (float*)d_out, nullptr, D);
}
